// DGCNNSeg_5609227289184
// MI455X (gfx1250) — matrix operand order repaired and checked by reading, not compiled, not run on hardware
//
#include <hip/hip_runtime.h>
#include <hip/hip_bf16.h>
#include <cstdint>

#define DEVINL __device__ __forceinline__

typedef __attribute__((ext_vector_type(16))) _Float16 v16h;
typedef __attribute__((ext_vector_type(8)))  float    v8f;

static constexpr int B   = 8;
static constexpr int N   = 2048;
static constexpr int KNN = 20;
static constexpr float NEG_SLOPE = 0.2f;

// ---------------------------------------------------------------------------
// WMMA fragment helpers (wave32, V_WMMA_F32_16X16X32_F16)
// Layouts used by these helpers (contiguous k order, A and B match; not the ISA table order):
//  A (16x32 f16): lane<16 holds row=lane, K=0..15;
//                 lane>=16 holds row=lane-16, K=16..31
//  B (32x16 f16): lanes 0-15 hold K=0..15 of col=lane, lanes16-31 K=16..31
//  C/D (16x16 f32): vgpr v, lane l -> (m = v + 8*(l>>4), n = l&15)
// ---------------------------------------------------------------------------
DEVINL v16h load_a_frag(const _Float16* __restrict__ src, int ld, int lane) {
  const int row  = lane & 15;
  const int koff = (lane >> 4) << 4;                // 0 or 16
  const _Float16* r = src + (size_t)row * ld;
  v16h a;
#pragma unroll
  for (int i = 0; i < 16; ++i) a[i] = r[koff + i];
  return a;
}

// B tile where B[k][n] = w[n*ldk + k] (row-major (O,K) weights, B = W^T);
// also used for gram X·X^T with w = X rows. Each lane reads 16 contiguous B
// halves -> lowers to 2x global_load_b128 (verified in round-1 disasm).
DEVINL v16h load_b_frag_t(const _Float16* __restrict__ w, int ldk, int lane) {
  const int col   = lane & 15;
  const int kbase = (lane >> 4) << 4;               // 0 or 16
  const _Float16* r = w + (size_t)col * ldk + kbase;
  v16h b;
#pragma unroll
  for (int i = 0; i < 16; ++i) b[i] = r[i];
  return b;
}

DEVINL v8f wmma_f16(v16h a, v16h b, v8f c) {
  return __builtin_amdgcn_wmma_f32_16x16x32_f16(
      /*neg_a=*/false, a, /*neg_b=*/false, b,
      /*c_mod=*/(short)0, c, /*reuse_a=*/false, /*reuse_b=*/false);
}

// ---------------------------------------------------------------------------
// kNN: per block, 16 query rows vs all N points. Gram via WMMA into LDS in
// 512-column chunks. Top-k: 4 scanner threads per row keep private sorted
// top-20 over disjoint column segments; 4-way sorted merge at the end.
// Self-distance 0 is included, matching jax.lax.top_k(-d, k).
// ---------------------------------------------------------------------------
template <int C>
__global__ __launch_bounds__(256) void knn_kernel(
    const _Float16* __restrict__ x, int ldx, int* __restrict__ idx) {
  constexpr int CH = 512;
  constexpr int NC = C / 32;
  constexpr int ST = 4;                 // scanner threads per row
  __shared__ float sdot[16][CH + 8];
  __shared__ float sxxc[CH];
  __shared__ float sxxq[16];
  __shared__ float cd[16 * ST][KNN];
  __shared__ int   ci[16 * ST][KNN];

  const int b    = blockIdx.x / (N / 16);
  const int qt   = blockIdx.x % (N / 16);
  const int tid  = threadIdx.x;
  const int lane = tid & 31;
  const int wave = tid >> 5;

  const _Float16* xb = x + (size_t)b * N * ldx;
  const int q0 = qt * 16;

  if (tid < 16) {
    float s = 0.f;
    const _Float16* r = xb + (size_t)(q0 + tid) * ldx;
    for (int c = 0; c < C; ++c) { float v = (float)r[c]; s += v * v; }
    sxxq[tid] = s;
  }

  v16h af[NC];
#pragma unroll
  for (int kc = 0; kc < NC; ++kc)
    af[kc] = load_a_frag(xb + (size_t)q0 * ldx + kc * 32, ldx, lane);

  float bd[KNN];
  int   bi[KNN];
#pragma unroll
  for (int i = 0; i < KNN; ++i) { bd[i] = 3.4e38f; bi[i] = 0; }

  const int srow = tid & 15;            // scanner row  (tid < 64)
  const int sseg = tid >> 4;            // scanner segment 0..3

  for (int ch = 0; ch < N / CH; ++ch) {
    const int j0 = ch * CH;
    __syncthreads();                    // prev scan done before overwrite
    for (int jj = tid; jj < CH; jj += 256) {
      float s = 0.f;
      const _Float16* r = xb + (size_t)(j0 + jj) * ldx;
      for (int c = 0; c < C; ++c) { float v = (float)r[c]; s += v * v; }
      sxxc[jj] = s;
    }
    for (int t32 = wave; t32 < CH / 16; t32 += 8) {
      v8f acc = {};
#pragma unroll
      for (int kc = 0; kc < NC; ++kc) {
        v16h bf = load_b_frag_t(xb + (size_t)(j0 + t32 * 16) * ldx + kc * 32,
                                ldx, lane);
        acc = wmma_f16(af[kc], bf, acc);
      }
      const int col = lane & 15, half = lane >> 4;
#pragma unroll
      for (int v = 0; v < 8; ++v)
        sdot[v + 8 * half][t32 * 16 + col] = acc[v];
    }
    __syncthreads();
    if (tid < 16 * ST) {
      const float xq = sxxq[srow];
      for (int jj = sseg * (CH / ST); jj < (sseg + 1) * (CH / ST); ++jj) {
        const float d = xq + sxxc[jj] - 2.f * sdot[srow][jj];
        if (d < bd[KNN - 1]) {
          int pos = KNN - 1;
          while (pos > 0 && bd[pos - 1] > d) {
            bd[pos] = bd[pos - 1]; bi[pos] = bi[pos - 1]; --pos;
          }
          bd[pos] = d; bi[pos] = j0 + jj;
        }
      }
    }
  }
  __syncthreads();
  if (tid < 16 * ST) {
#pragma unroll
    for (int i = 0; i < KNN; ++i) { cd[tid][i] = bd[i]; ci[tid][i] = bi[i]; }
  }
  __syncthreads();
  if (tid < 16) {
    float fd[KNN]; int fi[KNN];
#pragma unroll
    for (int i = 0; i < KNN; ++i) { fd[i] = cd[tid][i]; fi[i] = ci[tid][i]; }
    for (int s2 = 1; s2 < ST; ++s2) {
      const float* ld = cd[s2 * 16 + tid];
      const int*   li = ci[s2 * 16 + tid];
      for (int e = 0; e < KNN; ++e) {
        const float d = ld[e];
        if (d >= fd[KNN - 1]) break;    // lists are sorted ascending
        int pos = KNN - 1;
        while (pos > 0 && fd[pos - 1] > d) {
          fd[pos] = fd[pos - 1]; fi[pos] = fi[pos - 1]; --pos;
        }
        fd[pos] = d; fi[pos] = li[e];
      }
    }
    int* o = idx + ((size_t)b * N + q0 + tid) * KNN;
#pragma unroll
    for (int i = 0; i < KNN; ++i) o[i] = fi[i];
  }
}

// ---------------------------------------------------------------------------
// EdgeConv: 8 points x 20 neighbors = 160 edge rows per block (10 row tiles).
// Since s>0 and lrelu is monotone, max_k lrelu(s*h+t) = lrelu(s*max_k h + t):
// max-reduce raw WMMA outputs, activate once. Race-free ownership:
// each wave owns (column stripe, point-half); row 80 is a point boundary
// (4*20), so splitting the 10 row tiles in half splits points cleanly.
// A fragments are hoisted to the row-tile loop (loaded once per rt).
// ---------------------------------------------------------------------------
template <int CIN, int KPAD, int COUT>
__global__ __launch_bounds__(256) void edgeconv_kernel(
    const _Float16* __restrict__ x, int ldx, const int* __restrict__ idx,
    const _Float16* __restrict__ w, const float* __restrict__ sc,
    const float* __restrict__ sh, _Float16* __restrict__ out, int ldo) {
  constexpr int PPB  = 8;
  constexpr int ROWS = PPB * KNN;          // 160
  constexpr int RT   = ROWS / 16;          // 10
  constexpr int CT   = COUT / 16;
  constexpr int WPC  = (CT >= 8) ? 1 : (8 / CT);  // waves per column stripe
  constexpr int RTSEG = RT / WPC;
  __shared__ _Float16 ef[ROWS][KPAD];
  __shared__ float    mbuf[PPB][COUT];

  const int b    = blockIdx.x / (N / PPB);
  const int p0   = (blockIdx.x % (N / PPB)) * PPB;
  const int tid  = threadIdx.x;
  const int lane = tid & 31;
  const int wave = tid >> 5;

  for (int e = tid; e < ROWS * CIN; e += 256) {
    const int r = e / CIN, c = e % CIN;
    const int p = p0 + r / KNN;
    const int j = idx[((size_t)b * N + p) * KNN + (r % KNN)];
    const _Float16 xi = x[((size_t)b * N + p) * ldx + c];
    const _Float16 xj = x[((size_t)b * N + j) * ldx + c];
    ef[r][c]       = (_Float16)((float)xj - (float)xi);   // neigh - xi
    ef[r][CIN + c] = xi;                                  // xi
  }
  if (KPAD > 2 * CIN) {
    constexpr int PADC = KPAD - 2 * CIN;
    for (int e = tid; e < ROWS * PADC; e += 256)
      ef[e / PADC][2 * CIN + e % PADC] = (_Float16)0.f;
  }
  for (int e = tid; e < PPB * COUT; e += 256)
    mbuf[e / COUT][e % COUT] = -3.4e38f;
  __syncthreads();

  const int seg = wave / CT;               // 0 when CT>=8
  const int ct0 = (CT >= 8) ? wave : (wave % CT);
  for (int rt = seg * RTSEG; rt < (seg + 1) * RTSEG; ++rt) {
    v16h a[KPAD / 32];
#pragma unroll
    for (int kc = 0; kc < KPAD / 32; ++kc)
      a[kc] = load_a_frag(&ef[rt * 16][kc * 32], KPAD, lane);
    for (int ct = ct0; ct < CT; ct += 8) {
      v8f acc = {};
#pragma unroll
      for (int kc = 0; kc < KPAD / 32; ++kc) {
        v16h bf = load_b_frag_t(w + (size_t)(ct * 16) * KPAD + kc * 32, KPAD, lane);
        acc = wmma_f16(a[kc], bf, acc);
      }
      const int o    = ct * 16 + (lane & 15);
      const int half = lane >> 4;
#pragma unroll
      for (int h = 0; h < 2; ++h) {
        if (half == h) {                   // serialize halves: no LDS race
#pragma unroll
          for (int v = 0; v < 8; ++v) {
            const int row = rt * 16 + v + 8 * h;
            const int p   = row / KNN;
            mbuf[p][o] = fmaxf(mbuf[p][o], acc[v]);
          }
        }
      }
    }
  }
  __syncthreads();
  for (int e = tid; e < PPB * COUT; e += 256) {
    const int p = e / COUT, o = e % COUT;
    float h = mbuf[p][o] * sc[o] + sh[o];
    h = h >= 0.f ? h : NEG_SLOPE * h;
    out[((size_t)b * N + p0 + p) * ldo + o] = (_Float16)h;
  }
}

// ---------------------------------------------------------------------------
// Pointwise GEMM: 32 output rows (2 A tiles) per wave, 4 column tiles per
// accumulator group -> A loads amortized 4x, B loads 2x.
// ACT=true: f16 out with lrelu(s*h+t).  ACT=false (h3): f32 logits + bias.
// ---------------------------------------------------------------------------
template <int KD, int COUT, int COUTPAD, bool ACT>
__global__ __launch_bounds__(256) void pw_kernel(
    const _Float16* __restrict__ in, const _Float16* __restrict__ w,
    const float* __restrict__ sc, const float* __restrict__ sh,
    _Float16* __restrict__ outh, int ldo, float* __restrict__ outf) {
  constexpr int NCT = COUTPAD / 16;
  constexpr int CTG = (NCT >= 4) ? 4 : 1;
  const int lane  = threadIdx.x & 31;
  const int wave  = threadIdx.x >> 5;
  const int rbase = (blockIdx.x * 8 + wave) * 32;

  for (int ctg = 0; ctg < NCT; ctg += CTG) {
    v8f acc[2][CTG] = {};
    for (int kc = 0; kc < KD / 32; ++kc) {
      if (kc + 1 < KD / 32)
        __builtin_prefetch(in + (size_t)rbase * KD + (kc + 1) * 32, 0, 3);
      v16h a0 = load_a_frag(in + (size_t)rbase * KD + kc * 32, KD, lane);
      v16h a1 = load_a_frag(in + (size_t)(rbase + 16) * KD + kc * 32, KD, lane);
#pragma unroll
      for (int j = 0; j < CTG; ++j) {
        v16h bf = load_b_frag_t(w + (size_t)((ctg + j) * 16) * KD + kc * 32,
                                KD, lane);
        acc[0][j] = wmma_f16(a0, bf, acc[0][j]);
        acc[1][j] = wmma_f16(a1, bf, acc[1][j]);
      }
    }
    const int half = lane >> 4;
#pragma unroll
    for (int j = 0; j < CTG; ++j) {
      const int o = (ctg + j) * 16 + (lane & 15);
#pragma unroll
      for (int mt = 0; mt < 2; ++mt) {
#pragma unroll
        for (int v = 0; v < 8; ++v) {
          const int row = rbase + mt * 16 + v + 8 * half;
          if (ACT) {
            float h = acc[mt][j][v] * sc[o] + sh[o];
            h = h >= 0.f ? h : NEG_SLOPE * h;
            outh[(size_t)row * ldo + o] = (_Float16)h;
          } else {
            if (o < COUT) outf[(size_t)row * COUT + o] = acc[mt][j][v] + sh[o];
          }
        }
      }
    }
  }
}

// ---------------------------------------------------------------------------
// Small utility kernels
// ---------------------------------------------------------------------------
__global__ void cvt_weight_kernel(const float* __restrict__ src,
                                  _Float16* __restrict__ dst, int O, int C,
                                  int KPAD, int OPAD) {
  int e = blockIdx.x * 256 + threadIdx.x;
  if (e >= OPAD * KPAD) return;
  int o = e / KPAD, c = e % KPAD;
  dst[e] = (_Float16)((o < O && c < C) ? src[o * C + c] : 0.f);
}

__global__ void pad_xyz_kernel(const float* __restrict__ xyz,
                               _Float16* __restrict__ dst) {
  int e = blockIdx.x * 256 + threadIdx.x;
  if (e >= B * N * 32) return;
  int n = e / 32, c = e % 32;
  dst[e] = (c < 3) ? (_Float16)xyz[n * 3 + c] : (_Float16)0.f;
}

__global__ void globmax_kernel(const _Float16* __restrict__ x,
                               _Float16* __restrict__ g) {
  int e = blockIdx.x * 256 + threadIdx.x;   // B*1024
  if (e >= B * 1024) return;
  int b = e / 1024, o = e % 1024;
  float m = -3.4e38f;
  for (int n = 0; n < N; ++n)
    m = fmaxf(m, (float)x[((size_t)b * N + n) * 1024 + o]);
  g[e] = (_Float16)m;
}

__global__ void buildfin_kernel(const _Float16* __restrict__ loc,
                                const _Float16* __restrict__ glob,
                                _Float16* __restrict__ fin) {
  int e = blockIdx.x * 256 + threadIdx.x;   // B*N*1536
  if (e >= B * N * 1536) return;
  int row = e / 1536, c = e % 1536;
  int b = row / N;
  fin[e] = (c < 512) ? loc[(size_t)row * 512 + c]
                     : glob[b * 1024 + (c - 512)];
}

// ---------------------------------------------------------------------------
extern "C" void kernel_launch(void* const* d_in, const int* in_sizes, int n_in,
                              void* d_out, int out_size, void* d_ws,
                              size_t ws_size, hipStream_t stream) {
  (void)in_sizes; (void)n_in; (void)out_size; (void)ws_size;
  const float* xyz    = (const float*)d_in[0];
  const float* ec1_w  = (const float*)d_in[1];
  const float* ec1_s  = (const float*)d_in[2];
  const float* ec1_t  = (const float*)d_in[3];
  const float* ec2_w  = (const float*)d_in[4];
  const float* ec2_s  = (const float*)d_in[5];
  const float* ec2_t  = (const float*)d_in[6];
  const float* ec3_w  = (const float*)d_in[7];
  const float* ec3_s  = (const float*)d_in[8];
  const float* ec3_t  = (const float*)d_in[9];
  const float* ec4_w  = (const float*)d_in[10];
  const float* ec4_s  = (const float*)d_in[11];
  const float* ec4_t  = (const float*)d_in[12];
  const float* fuse_w = (const float*)d_in[13];
  const float* fuse_s = (const float*)d_in[14];
  const float* fuse_t = (const float*)d_in[15];
  const float* emb_w  = (const float*)d_in[16];
  const float* emb_s  = (const float*)d_in[17];
  const float* emb_t  = (const float*)d_in[18];
  const float* h1_w   = (const float*)d_in[19];
  const float* h1_s   = (const float*)d_in[20];
  const float* h1_t   = (const float*)d_in[21];
  const float* h2_w   = (const float*)d_in[22];
  const float* h2_s   = (const float*)d_in[23];
  const float* h2_t   = (const float*)d_in[24];
  const float* h3_w   = (const float*)d_in[25];
  const float* h3_b   = (const float*)d_in[26];
  float* logits = (float*)d_out;

  char* ws = (char*)d_ws;
  size_t off = 0;
  auto alloc = [&](size_t bytes) -> char* {
    char* p = ws + off;
    off += (bytes + 255) & ~(size_t)255;
    return p;
  };
  const size_t M = (size_t)B * N;
  _Float16* xyzh  = (_Float16*)alloc(M * 32 * 2);
  int*      idxb  = (int*)alloc(M * KNN * 4);
  _Float16* xcat  = (_Float16*)alloc(M * 512 * 2);   // x1|x2|x3|x4 slices
  _Float16* xloc  = (_Float16*)alloc(M * 512 * 2);
  _Float16* xemb  = (_Float16*)alloc(M * 1024 * 2);
  _Float16* xglob = (_Float16*)alloc((size_t)B * 1024 * 2);
  _Float16* xfin  = (_Float16*)alloc(M * 1536 * 2);
  _Float16* h1o   = (_Float16*)alloc(M * 256 * 2);
  _Float16* h2o   = (_Float16*)alloc(M * 256 * 2);
  _Float16* wec1  = (_Float16*)alloc(64 * 32 * 2);
  _Float16* wec2  = (_Float16*)alloc(64 * 128 * 2);
  _Float16* wec3  = (_Float16*)alloc(128 * 128 * 2);
  _Float16* wec4  = (_Float16*)alloc(256 * 256 * 2);
  _Float16* wfuse = (_Float16*)alloc(512 * 512 * 2);
  _Float16* wemb  = (_Float16*)alloc(1024 * 512 * 2);
  _Float16* wh1   = (_Float16*)alloc(256 * 1536 * 2);
  _Float16* wh2   = (_Float16*)alloc(256 * 256 * 2);
  _Float16* wh3   = (_Float16*)alloc(16 * 256 * 2);

  auto cvtw = [&](const float* s, _Float16* d, int O, int C, int KP, int OP) {
    int tot = OP * KP;
    cvt_weight_kernel<<<(tot + 255) / 256, 256, 0, stream>>>(s, d, O, C, KP, OP);
  };
  cvtw(ec1_w,  wec1,  64,   6,   32,   64);
  cvtw(ec2_w,  wec2,  64,   128, 128,  64);
  cvtw(ec3_w,  wec3,  128,  128, 128,  128);
  cvtw(ec4_w,  wec4,  256,  256, 256,  256);
  cvtw(fuse_w, wfuse, 512,  512, 512,  512);
  cvtw(emb_w,  wemb,  1024, 512, 512,  1024);
  cvtw(h1_w,   wh1,   256, 1536, 1536, 256);
  cvtw(h2_w,   wh2,   256,  256, 256,  256);
  cvtw(h3_w,   wh3,   13,   256, 256,  16);

  pad_xyz_kernel<<<(B * N * 32 + 255) / 256, 256, 0, stream>>>(xyz, xyzh);

  const int gknn = B * (N / 16);       // 1024 blocks
  const int gec  = B * (N / 8);        // 2048 blocks

  // layer 1: xyz(pad32) -> x1 (64ch, cols 0..63 of xcat)
  knn_kernel<32><<<gknn, 256, 0, stream>>>(xyzh, 32, idxb);
  edgeconv_kernel<3, 32, 64><<<gec, 256, 0, stream>>>(
      xyzh, 32, idxb, wec1, ec1_s, ec1_t, xcat + 0, 512);
  // layer 2: x1 -> x2 (64ch, cols 64..127)
  knn_kernel<64><<<gknn, 256, 0, stream>>>(xcat + 0, 512, idxb);
  edgeconv_kernel<64, 128, 64><<<gec, 256, 0, stream>>>(
      xcat + 0, 512, idxb, wec2, ec2_s, ec2_t, xcat + 64, 512);
  // layer 3: x2 -> x3 (128ch, cols 128..255)
  knn_kernel<64><<<gknn, 256, 0, stream>>>(xcat + 64, 512, idxb);
  edgeconv_kernel<64, 128, 128><<<gec, 256, 0, stream>>>(
      xcat + 64, 512, idxb, wec3, ec3_s, ec3_t, xcat + 128, 512);
  // layer 4: x3 -> x4 (256ch, cols 256..511)
  knn_kernel<128><<<gknn, 256, 0, stream>>>(xcat + 128, 512, idxb);
  edgeconv_kernel<128, 256, 256><<<gec, 256, 0, stream>>>(
      xcat + 128, 512, idxb, wec4, ec4_s, ec4_t, xcat + 256, 512);

  const int gpw = (int)(M / 256);      // 64 blocks (32 rows per wave)
  // fuse: 512 -> 512
  pw_kernel<512, 512, 512, true><<<gpw, 256, 0, stream>>>(
      xcat, wfuse, fuse_s, fuse_t, xloc, 512, nullptr);
  // emb: 512 -> 1024
  pw_kernel<512, 1024, 1024, true><<<gpw, 256, 0, stream>>>(
      xloc, wemb, emb_s, emb_t, xemb, 1024, nullptr);
  // global max over points
  globmax_kernel<<<(B * 1024 + 255) / 256, 256, 0, stream>>>(xemb, xglob);
  // x_fin = [x_local | broadcast(x_glob)]
  buildfin_kernel<<<(int)((M * 1536 + 255) / 256), 256, 0, stream>>>(
      xloc, xglob, xfin);
  // heads
  pw_kernel<1536, 256, 256, true><<<gpw, 256, 0, stream>>>(
      xfin, wh1, h1_s, h1_t, h1o, 256, nullptr);
  pw_kernel<256, 256, 256, true><<<gpw, 256, 0, stream>>>(
      h1o, wh2, h2_s, h2_t, h2o, 256, nullptr);
  pw_kernel<256, 13, 16, false><<<gpw, 256, 0, stream>>>(
      h2o, wh3, nullptr, h3_b, nullptr, 0, logits);
}
